// SocialLSTMClassifier_73486890434911
// MI455X (gfx1250) — matrix operand order repaired and checked by reading, not compiled, not run on hardware
//
#include <hip/hip_runtime.h>
#include <hip/hip_bf16.h>
#include <stdint.h>

// ---- problem constants (from reference) ----
constexpr int kObs = 15;
constexpr int kN   = 50000;   // neighbors (multiple of 16 -> full WMMA tiles)
constexpr int kH   = 64;      // hidden
constexpr int kG   = 4 * kH;  // 256 gates
constexpr int kCells = 16;    // 4x4 social grid
constexpr int kWPad  = 72;    // LDS row stride (halfs) for W_hh: conflict-free b128 reads

typedef _Float16 v16h __attribute__((ext_vector_type(16)));
typedef _Float16 v8h  __attribute__((ext_vector_type(8)));
typedef _Float16 v4h  __attribute__((ext_vector_type(4)));
typedef float    v8f  __attribute__((ext_vector_type(8)));

// ---- fast activations: gfx1250 hardware v_tanh_f32 (confirmed in round 2 asm);
// ---- fall back to exp + raw v_rcp_f32 if builtin is absent.
#if defined(__has_builtin)
#  if __has_builtin(__builtin_amdgcn_tanhf)
#    define HAS_HW_TANH 1
#  endif
#endif

__device__ __forceinline__ float fast_rcp(float x) { return __builtin_amdgcn_rcpf(x); }

#ifdef HAS_HW_TANH
__device__ __forceinline__ float tanh_(float x) { return __builtin_amdgcn_tanhf(x); }
__device__ __forceinline__ float sig_(float x)  { return fmaf(0.5f, __builtin_amdgcn_tanhf(0.5f * x), 0.5f); }
#else
__device__ __forceinline__ float tanh_(float x) {
    float t = __expf(-2.0f * fabsf(x));
    return copysignf((1.0f - t) * fast_rcp(1.0f + t), x);
}
__device__ __forceinline__ float sig_(float x) { return fast_rcp(1.0f + __expf(-x)); }
#endif

// ---------------------------------------------------------------------------
// init: zero recurrent state + social accumulator, convert W_hh -> f16,
// build the f16 (W_ih | bias) fragment table (entry kG is all-zero, used by
// lanes 16-31 so the x/bias WMMA needs no masking).
// ---------------------------------------------------------------------------
__global__ void init_kernel(const float* __restrict__ Whh,
                            const float* __restrict__ W_ih,
                            const float* __restrict__ b_ih,
                            const float* __restrict__ b_hh,
                            _Float16* __restrict__ Whh16,
                            _Float16* __restrict__ xbTab,
                            _Float16* __restrict__ hB,
                            float* __restrict__ cB,
                            float* __restrict__ social,
                            float* __restrict__ hT,
                            float* __restrict__ cT) {
    int idx = blockIdx.x * blockDim.x + threadIdx.x;
    if (idx < kN * kH) { hB[idx] = (_Float16)0.0f; cB[idx] = 0.0f; }
    if (idx < kG * kH) Whh16[idx] = (_Float16)Whh[idx];
    if (idx < kG) {
        xbTab[idx * 4 + 0] = (_Float16)W_ih[2 * idx];
        xbTab[idx * 4 + 1] = (_Float16)W_ih[2 * idx + 1];
        xbTab[idx * 4 + 2] = (_Float16)(b_ih[idx] + b_hh[idx]);
        xbTab[idx * 4 + 3] = (_Float16)0.0f;
    }
    if (idx == kG) {
        xbTab[idx * 4 + 0] = (_Float16)0.0f; xbTab[idx * 4 + 1] = (_Float16)0.0f;
        xbTab[idx * 4 + 2] = (_Float16)0.0f; xbTab[idx * 4 + 3] = (_Float16)0.0f;
    }
    if (idx < kCells * kH) social[idx] = 0.0f;
    if (idx < kH) { hT[idx] = 0.0f; cT[idx] = 0.0f; }
}

// ---------------------------------------------------------------------------
// Per-step neighbor LSTM (WMMA) + fused social pooling.
// One wave32 handles 16 neighbors. gates[16x256] = [h | x | 1] @ [Whh^T ;
// W_ih^T ; b]^T entirely in v_wmma_f32_16x16x32_f16 (12 WMMA per 16-wide k
// slice: 4 x/bias + 8 recurrent). W_hh lives in LDS (padded rows, conflict-
// free ds_load_b128); c-state lives in WMMA-fragment-major layout (b128 IO).
// ---------------------------------------------------------------------------
__global__ __launch_bounds__(256)
void neighbor_step_kernel(const float* __restrict__ tgt,
                          const float* __restrict__ oth,
                          const int*   __restrict__ mask,
                          const _Float16* __restrict__ Whh16,
                          const _Float16* __restrict__ xbTab,
                          _Float16* __restrict__ hB,
                          float* __restrict__ cB,
                          float* __restrict__ social,
                          int t) {
    __shared__ _Float16 whhS[kG * kWPad];   // 36 KB staged weights
    __shared__ float sPool[kCells * kH];    // 4 KB social accumulator
    const int tid  = threadIdx.x;
    const int lane = tid & 31;
    const int wave = tid >> 5;

    // stage W_hh (f16) into LDS with padded rows: 2048 16B chunks, coalesced
    for (int i = tid; i < kG * kH / 8; i += 256) {
        const int row = i >> 3, c8 = (i & 7) * 8;
        *(v8h*)(whhS + row * kWPad + c8) = *(const v8h*)(Whh16 + row * kH + c8);
    }
    for (int i = tid; i < kCells * kH; i += 256) sPool[i] = 0.0f;
    __syncthreads();

    const int nTiles = kN / 16;                 // 3125
    const int tile   = blockIdx.x * 8 + wave;   // wave-uniform => EXEC stays full
    if (tile < nTiles) {
        const int n0   = tile * 16;
        const int lmod = lane & 15;
        const int hi   = (lane >> 4) & 1;

        // ---- per-neighbor scalars (lane l and l+16 mirror neighbor n0+l) ----
        const int   nl   = n0 + lmod;
        const float ref0 = tgt[t * 2 + 0];
        const float ref1 = tgt[t * 2 + 1];
        const float x0v  = oth[((size_t)t * kN + nl) * 2 + 0];
        const float x1v  = oth[((size_t)t * kN + nl) * 2 + 1];
        const int   mv   = mask[(size_t)t * kN + nl];
        const float r0 = x0v - ref0, r1 = x1v - ref1;
        const int cx = (int)truncf(r0) + 2;        // cw = ch = 1.0, hx = hy = 2
        const int cy = (int)truncf(r1) + 2;
        const bool ok = (fabsf(r0) <= 2.0f) && (fabsf(r1) <= 2.0f) &&
                        (cx >= 0) && (cx < 4) && (cy >= 0) && (cy < 4) && (mv != 0);
        const int cellv = ok ? (cy * 4 + cx) : -1;

        // grid cells of the 8 neighbor rows this lane post-processes (hoisted)
        int cell_e[8];
        #pragma unroll
        for (int e = 0; e < 8; ++e) cell_e[e] = __shfl(cellv, e + 8 * hi, 32);

        // ---- A fragments (contiguous k = 16*hi + s per lane half), K = 0..63 ----
        const _Float16* hrow = hB + (size_t)nl * kH;
        const int klo = hi * 16;
        v8h a0l = *(const v8h*)(hrow + klo);
        v8h a0h = *(const v8h*)(hrow + klo + 8);
        v8h a1l = *(const v8h*)(hrow + 32 + klo);
        v8h a1h = *(const v8h*)(hrow + 32 + klo + 8);
        v16h a0 = __builtin_shufflevector(a0l, a0h, 0,1,2,3,4,5,6,7,8,9,10,11,12,13,14,15);
        v16h a1 = __builtin_shufflevector(a1l, a1h, 0,1,2,3,4,5,6,7,8,9,10,11,12,13,14,15);

        // A' for the folded x/bias rank-3 update: row m = [x0, x1, 1, 0...0]
        v16h ax = {};
        if (!hi) {
            ax[0] = (_Float16)x0v;
            ax[1] = (_Float16)x1v;
            ax[2] = (_Float16)1.0f;
        }

        const v4h z4 = {};
        #pragma unroll
        for (int kb = 0; kb < 4; ++kb) {            // gate column slice kb*16..kb*16+15
            const int gI = kb * 16 + lmod;          // i/f/g/o gate rows: gI, +64, +128, +192
            // B' fragments from the f16 (W_ih | bias) table; lanes 16-31 read the zero row
            v4h tI = *(const v4h*)(xbTab + 4 * (hi ? kG : gI));
            v4h tF = *(const v4h*)(xbTab + 4 * (hi ? kG : gI + 64));
            v4h tG = *(const v4h*)(xbTab + 4 * (hi ? kG : gI + 128));
            v4h tO = *(const v4h*)(xbTab + 4 * (hi ? kG : gI + 192));
            v16h bxI = __builtin_shufflevector(tI, z4, 0,1,2,3,4,4,4,4,4,4,4,4,4,4,4,4);
            v16h bxF = __builtin_shufflevector(tF, z4, 0,1,2,3,4,4,4,4,4,4,4,4,4,4,4,4);
            v16h bxG = __builtin_shufflevector(tG, z4, 0,1,2,3,4,4,4,4,4,4,4,4,4,4,4,4);
            v16h bxO = __builtin_shufflevector(tO, z4, 0,1,2,3,4,4,4,4,4,4,4,4,4,4,4,4);

            v8f ai = {}; v8f af = {}; v8f ag = {}; v8f ao = {};
            ai = __builtin_amdgcn_wmma_f32_16x16x32_f16(false, ax, false, bxI, (short)0, ai, false, false);
            af = __builtin_amdgcn_wmma_f32_16x16x32_f16(false, ax, false, bxF, (short)0, af, false, false);
            ag = __builtin_amdgcn_wmma_f32_16x16x32_f16(false, ax, false, bxG, (short)0, ag, false, false);
            ao = __builtin_amdgcn_wmma_f32_16x16x32_f16(false, ax, false, bxO, (short)0, ao, false, false);
            #pragma unroll
            for (int kh = 0; kh < 2; ++kh) {        // K halves 0..31 / 32..63
                // B (32x16): lane = column j (row j of W_hh), lanes 0-15 hold
                // K0..15, lanes 16-31 hold K16..31 -> 32B contiguous per lane.
                // Padded row stride (36 words) makes both ds_load_b128 sweeps
                // hit all 64 banks exactly once (bases = 4*l mod 64).
                const int boff = kh * 32 + hi * 16;
                const _Float16* bpI = whhS + (gI      ) * kWPad + boff;
                const _Float16* bpF = whhS + (gI +  64) * kWPad + boff;
                const _Float16* bpG = whhS + (gI + 128) * kWPad + boff;
                const _Float16* bpO = whhS + (gI + 192) * kWPad + boff;
                v16h bI = __builtin_shufflevector(*(const v8h*)bpI, *(const v8h*)(bpI + 8),
                                                  0,1,2,3,4,5,6,7,8,9,10,11,12,13,14,15);
                v16h bF = __builtin_shufflevector(*(const v8h*)bpF, *(const v8h*)(bpF + 8),
                                                  0,1,2,3,4,5,6,7,8,9,10,11,12,13,14,15);
                v16h bG = __builtin_shufflevector(*(const v8h*)bpG, *(const v8h*)(bpG + 8),
                                                  0,1,2,3,4,5,6,7,8,9,10,11,12,13,14,15);
                v16h bO = __builtin_shufflevector(*(const v8h*)bpO, *(const v8h*)(bpO + 8),
                                                  0,1,2,3,4,5,6,7,8,9,10,11,12,13,14,15);
                v16h av = kh ? a1 : a0;
                ai = __builtin_amdgcn_wmma_f32_16x16x32_f16(false, av, false, bI, (short)0, ai, false, false);
                af = __builtin_amdgcn_wmma_f32_16x16x32_f16(false, av, false, bF, (short)0, af, false, false);
                ag = __builtin_amdgcn_wmma_f32_16x16x32_f16(false, av, false, bG, (short)0, ag, false, false);
                ao = __builtin_amdgcn_wmma_f32_16x16x32_f16(false, av, false, bO, (short)0, ao, false, false);
            }
            const int k = kb * 16 + lmod;           // hidden index handled by this lane
            // c-state in fragment-major layout: per (tile,kb), lane-contiguous
            // 8 floats -> two coalesced b128 each way.
            const size_t cbase = (((size_t)tile * 4 + kb) * 32 + lane) * 8;
            v8f cold8 = *(const v8f*)(cB + cbase);
            v8f cnew8;
            #pragma unroll
            for (int e = 0; e < 8; ++e) {
                const int m = e + 8 * hi;           // D: VGPR e -> row e (lanes<16) / e+8
                const float gi = ai[e], gf = af[e], gg = ag[e], go = ao[e];
                const float cnew = sig_(gf) * cold8[e] + sig_(gi) * tanh_(gg);
                const float hnew = sig_(go) * tanh_(cnew);
                cnew8[e] = cnew;
                hB[(size_t)(n0 + m) * kH + k] = (_Float16)hnew;  // row-major for next A
                const int cm = cell_e[e];
                if (cm >= 0) atomicAdd(&sPool[cm * kH + k], hnew);
            }
            *(v8f*)(cB + cbase) = cnew8;
        }
    }
    __syncthreads();
    for (int i = tid; i < kCells * kH; i += 256) {
        const float v = sPool[i];
        if (v != 0.0f) atomicAdd(&social[i], v);
    }
}

// ---------------------------------------------------------------------------
// Per-step target LSTM + social MLP (tiny, one block). Consumes `social`,
// then zeroes it for the next step. Writes final output at t == kObs-1.
// ---------------------------------------------------------------------------
__global__ __launch_bounds__(256)
void target_step_kernel(const float* __restrict__ tgt,
                        const float* __restrict__ W_ih, const float* __restrict__ b_ih,
                        const float* __restrict__ W_hh, const float* __restrict__ b_hh,
                        const float* __restrict__ W1,   const float* __restrict__ b1,
                        const float* __restrict__ W2,   const float* __restrict__ b2,
                        const float* __restrict__ Wc,   const float* __restrict__ bc,
                        float* __restrict__ hT, float* __restrict__ cT,
                        float* __restrict__ social,
                        float* __restrict__ out, int t) {
    __shared__ float hs[kH];
    __shared__ float gates[kG];
    __shared__ float sv[kCells * kH];
    __shared__ float hn[kH];
    __shared__ float ctx1[kH];
    __shared__ float comb[kH];
    const int tid = threadIdx.x;

    if (tid < kH) hs[tid] = hT[tid];
    for (int i = tid; i < kCells * kH; i += 256) sv[i] = social[i];
    __syncthreads();

    const float x0 = tgt[t * 2 + 0], x1 = tgt[t * 2 + 1];
    {   // gates = x @ W_ih^T + b_ih + h @ W_hh^T + b_hh   (256 threads, 1 gate each)
        float acc = b_ih[tid] + b_hh[tid] + x0 * W_ih[2 * tid] + x1 * W_ih[2 * tid + 1];
        const float* w = W_hh + (size_t)tid * kH;
        #pragma unroll 8
        for (int k = 0; k < kH; ++k) acc += w[k] * hs[k];
        gates[tid] = acc;
    }
    __syncthreads();
    if (tid < kH) {
        const float gi = gates[tid], gf = gates[64 + tid];
        const float gg = gates[128 + tid], go = gates[192 + tid];
        const float cnew = sig_(gf) * cT[tid] + sig_(gi) * tanh_(gg);
        const float h_   = sig_(go) * tanh_(cnew);
        cT[tid] = cnew; hT[tid] = h_; hn[tid] = h_;
    }
    if (tid < kH) {      // ctx1 = relu(sv @ W1^T + b1)
        float acc = b1[tid];
        const float* w = W1 + (size_t)tid * (kCells * kH);
        #pragma unroll 8
        for (int p = 0; p < kCells * kH; ++p) acc += w[p] * sv[p];
        ctx1[tid] = fmaxf(acc, 0.0f);
    }
    __syncthreads();
    if (tid < kH) {      // combined = h_t + ctx1 @ W2^T + b2
        float acc = b2[tid];
        const float* w = W2 + (size_t)tid * kH;
        #pragma unroll 8
        for (int k = 0; k < kH; ++k) acc += w[k] * ctx1[k];
        comb[tid] = hn[tid] + acc;
    }
    __syncthreads();
    for (int i = tid; i < kCells * kH; i += 256) social[i] = 0.0f;  // ready for next step
    if (t == kObs - 1 && tid < 2) {
        float acc = bc[tid];
        for (int j = 0; j < kH; ++j) acc += Wc[tid * kH + j] * comb[j];
        out[tid] = acc;
    }
}

// ---------------------------------------------------------------------------
extern "C" void kernel_launch(void* const* d_in, const int* in_sizes, int n_in,
                              void* d_out, int out_size, void* d_ws, size_t ws_size,
                              hipStream_t stream) {
    (void)in_sizes; (void)n_in; (void)out_size; (void)ws_size;
    const float* tgt  = (const float*)d_in[0];
    const float* oth  = (const float*)d_in[1];
    const int*   mask = (const int*)d_in[2];
    const float* W_ih = (const float*)d_in[3];
    const float* b_ih = (const float*)d_in[4];
    const float* W_hh = (const float*)d_in[5];
    const float* b_hh = (const float*)d_in[6];
    const float* W1   = (const float*)d_in[7];
    const float* b1   = (const float*)d_in[8];
    const float* W2   = (const float*)d_in[9];
    const float* b2   = (const float*)d_in[10];
    const float* Wc   = (const float*)d_in[11];
    const float* bc   = (const float*)d_in[12];
    float* out = (float*)d_out;

    // workspace carve-out (256B aligned): ~19.3 MB total
    char* p = (char*)d_ws;
    auto alloc = [&](size_t bytes) { char* r = p; p += (bytes + 255) & ~(size_t)255; return r; };
    _Float16* hB     = (_Float16*)alloc((size_t)kN * kH * sizeof(_Float16)); // 6.4 MB, row-major
    float*    cB     = (float*)   alloc((size_t)kN * kH * sizeof(float));    // 12.8 MB, fragment-major
    float*    social = (float*)   alloc(kCells * kH * sizeof(float));
    float*    hT     = (float*)   alloc(kH * sizeof(float));
    float*    cT     = (float*)   alloc(kH * sizeof(float));
    _Float16* Whh16  = (_Float16*)alloc((size_t)kG * kH * sizeof(_Float16)); // 32 KB
    _Float16* xbTab  = (_Float16*)alloc((size_t)(kG + 1) * 4 * sizeof(_Float16));

    const int initBlocks = (kN * kH + 255) / 256;
    init_kernel<<<initBlocks, 256, 0, stream>>>(W_hh, W_ih, b_ih, b_hh,
                                                Whh16, xbTab, hB, cB, social, hT, cT);

    const int nTiles  = kN / 16;            // 3125 full 16-neighbor tiles
    const int nBlocks = (nTiles + 7) / 8;   // 8 waves per block
    for (int t = 0; t < kObs; ++t) {
        neighbor_step_kernel<<<nBlocks, 256, 0, stream>>>(
            tgt, oth, mask, Whh16, xbTab, hB, cB, social, t);
        target_step_kernel<<<1, 256, 0, stream>>>(
            tgt, W_ih, b_ih, W_hh, b_hh, W1, b1, W2, b2, Wc, bc, hT, cT, social, out, t);
    }
}
